// MambaForModularArithmetic_53824530153927
// MI455X (gfx1250) — compile-verified, matrix operand order checked by reading, not run on hardware
//
#include <hip/hip_runtime.h>
#include <hip/hip_bf16.h>

// Problem constants (match reference)
#define BB    8
#define LSEQ  256
#define DMOD  1024
#define DINN  2048
#define NST   16
#define KCV   4
#define NLAY  4
#define DTRK  128
#define PHEAD 97
#define XPD   (DTRK + 2 * NST)   // 160

typedef __attribute__((ext_vector_type(16))) __bf16 v16bf;
typedef __attribute__((ext_vector_type(8)))  float  v8f;

// ---------------------------------------------------------------- utilities
__global__ void k_f32_to_bf16(const float* __restrict__ s,
                              __hip_bfloat16* __restrict__ d, long n) {
    long i = (long)blockIdx.x * blockDim.x + threadIdx.x;
    long stride = (long)gridDim.x * blockDim.x;
    for (; i < n; i += stride) d[i] = __float2bfloat16(s[i]);
}

__global__ void k_embed(const int* __restrict__ tok,
                        const float* __restrict__ te,
                        const float* __restrict__ pe,
                        float* __restrict__ h) {
    int bl = blockIdx.x;                 // 0..B*L-1
    int t  = bl % LSEQ;
    int v  = tok[bl];
    const float* terow = te + (long)v * DMOD;
    const float* perow = pe + (long)t * DMOD;
    float* hrow = h + (long)bl * DMOD;
    for (int i = threadIdx.x; i < DMOD; i += blockDim.x)
        hrow[i] = terow[i] + perow[i];
}

__global__ void k_layernorm(const float* __restrict__ x,
                            const float* __restrict__ w,
                            const float* __restrict__ b,
                            float* __restrict__ of,            // nullable
                            __hip_bfloat16* __restrict__ ob,   // nullable
                            int dim) {
    __shared__ float red[256];
    int row = blockIdx.x;
    int tid = threadIdx.x;
    const float* xr = x + (long)row * dim;

    float s = 0.f;
    for (int i = tid; i < dim; i += blockDim.x) s += xr[i];
    red[tid] = s; __syncthreads();
    for (int o = 128; o; o >>= 1) { if (tid < o) red[tid] += red[tid + o]; __syncthreads(); }
    float mean = red[0] / dim;
    __syncthreads();

    float v = 0.f;
    for (int i = tid; i < dim; i += blockDim.x) { float d0 = xr[i] - mean; v += d0 * d0; }
    red[tid] = v; __syncthreads();
    for (int o = 128; o; o >>= 1) { if (tid < o) red[tid] += red[tid + o]; __syncthreads(); }
    float rstd = rsqrtf(red[0] / dim + 1e-5f);

    for (int i = tid; i < dim; i += blockDim.x) {
        float o = (xr[i] - mean) * rstd * w[i] + b[i];
        if (of) of[(long)row * dim + i] = o;
        if (ob) ob[(long)row * dim + i] = __float2bfloat16(o);
    }
}

// ------------------------------------------------ bf16 WMMA GEMM: C = A * W^T
// A: M x K (row-major, lda),  W: N x K (row-major, ldw),  C: M x N (ldc, f32)
// Each wave32 computes a 16x64 output strip (4 accumulators): the A-fragment is
// loaded once per 32-wide K step and reused across 4 v_wmma_f32_16x16x32_bf16.
// Fragment striping used here (same for A and W): lane%16 = row, lane/16 selects the 16-wide K-half.
#define WMMA_BF16(ACC, AF, BF) \
    ACC = __builtin_amdgcn_wmma_f32_16x16x32_bf16(false, (AF), false, (BF), (short)0, ACC, false, false)

__global__ void __launch_bounds__(128)
k_gemm_bf16(const __hip_bfloat16* __restrict__ A, int lda,
            const __hip_bfloat16* __restrict__ W, int ldw,
            float* __restrict__ C, int ldc,
            int M, int N, int Kd) {
    int lane = threadIdx.x;                        // 0..31
    int r    = lane & 15;
    int half = lane >> 4;
    int nTiles = N >> 4;                           // N is a multiple of 16
    int tn0    = (blockIdx.x * blockDim.y + threadIdx.y) * 4; // first N-tile
    if (tn0 >= nTiles) return;
    int tileM = blockIdx.y;

    const __hip_bfloat16* Ar = A + (size_t)(tileM * 16 + r) * lda + half * 16;
    const __hip_bfloat16* Wr[4];
#pragma unroll
    for (int j = 0; j < 4; ++j) {
        int tn = tn0 + j;
        if (tn >= nTiles) tn = nTiles - 1;         // clamp: loads stay in-bounds
        Wr[j] = W + (size_t)(tn * 16 + r) * ldw + half * 16;
    }

    v8f acc0 = {0.f,0.f,0.f,0.f,0.f,0.f,0.f,0.f};
    v8f acc1 = acc0, acc2 = acc0, acc3 = acc0;

    for (int k0 = 0; k0 < Kd; k0 += 32) {
        v16bf a  = *reinterpret_cast<const v16bf*>(Ar + k0);
        if (k0 + 256 < Kd) __builtin_prefetch(Ar + k0 + 256, 0, 3); // global_prefetch_b8
        v16bf b0 = *reinterpret_cast<const v16bf*>(Wr[0] + k0);
        v16bf b1 = *reinterpret_cast<const v16bf*>(Wr[1] + k0);
        v16bf b2 = *reinterpret_cast<const v16bf*>(Wr[2] + k0);
        v16bf b3 = *reinterpret_cast<const v16bf*>(Wr[3] + k0);
        WMMA_BF16(acc0, a, b0);
        WMMA_BF16(acc1, a, b1);
        WMMA_BF16(acc2, a, b2);
        WMMA_BF16(acc3, a, b3);
    }

    // C layout: VGPR v -> row v (lanes 0-15) / row v+8 (lanes 16-31), col = lane%16
    float* Cb = C + (size_t)(tileM * 16 + half * 8) * ldc + r;
#pragma unroll
    for (int j = 0; j < 4; ++j) {
        if (tn0 + j >= nTiles) break;              // wave-uniform guard
        float* Cj = Cb + (size_t)(tn0 + j) * 16;
        v8f acc = (j == 0) ? acc0 : (j == 1) ? acc1 : (j == 2) ? acc2 : acc3;
#pragma unroll
        for (int v = 0; v < 8; ++v) Cj[(size_t)v * ldc] = acc[v];
    }
}

// ------------------------------------------- causal depthwise conv (K=4)+silu
__global__ void k_conv_silu(const float* __restrict__ xz,
                            const float* __restrict__ cw,
                            const float* __restrict__ cb,
                            float* __restrict__ xf,
                            __hip_bfloat16* __restrict__ xb) {
    long idx = (long)blockIdx.x * blockDim.x + threadIdx.x;   // over B*L*DI
    if (idx >= (long)BB * LSEQ * DINN) return;
    int d   = (int)(idx % DINN);
    long bl = idx / DINN;
    int t   = (int)(bl % LSEQ);
    long base = bl * (2L * DINN) + d;     // x-branch lives in cols [0,DI)
    float acc = cb[d];
#pragma unroll
    for (int k = 0; k < KCV; ++k) {
        int tt = t - (KCV - 1) + k;
        if (tt >= 0) acc += cw[d * KCV + k] * xz[base + (long)(tt - t) * (2L * DINN)];
    }
    float s = acc / (1.f + __expf(-acc));             // silu
    xf[idx] = s;
    xb[idx] = __float2bfloat16(s);
}

__global__ void k_softplus_bias(float* __restrict__ d, const float* __restrict__ b,
                                long n, int dim) {
    long i = (long)blockIdx.x * blockDim.x + threadIdx.x;
    if (i >= n) return;
    float x = d[i] + b[i % dim];
    d[i] = (x > 20.f) ? x : log1pf(__expf(x));
}

// ------------------------------------------------------ selective SSM scan
// grid (DI/256, B); each thread owns one channel d with 16-wide state in regs.
// B/C vectors for step t+1 are prefetched into LDS with the CDNA5 async
// global->LDS DMA path (ASYNCcnt) while step t is being computed.
__global__ void k_ssm_scan(const float* __restrict__ delta,
                           const float* __restrict__ xbf,
                           const float* __restrict__ xp,
                           const float* __restrict__ Alog,
                           const float* __restrict__ Dp,
                           float* __restrict__ y) {
    __shared__ float sBC[2][2 * NST];
    int tid = threadIdx.x;
    int b   = blockIdx.y;
    int d   = blockIdx.x * blockDim.x + tid;

    float an[NST], hst[NST];
#pragma unroll
    for (int n = 0; n < NST; ++n) {
        an[n]  = -__expf(Alog[(long)d * NST + n]);
        hst[n] = 0.f;
    }
    float dpar = Dp[d];

    // kick off async copy of B/C for t=0 into buffer 0
    if (tid < 2 * NST) {
        const float* g = xp + (long)b * LSEQ * XPD + DTRK + tid;
        unsigned lds = (unsigned)(size_t)(const void*)&sBC[0][tid];
        asm volatile("global_load_async_to_lds_b32 %0, %1, off"
                     :: "v"(lds), "v"(g) : "memory");
    }

    for (int t = 0; t < LSEQ; ++t) {
        int buf = t & 1;
        if (tid < 2 * NST && (t + 1) < LSEQ) {
            // prefetch next step's B/C into the other buffer, then retire
            // only the older async load (in-order completion => cnt<=1).
            const float* g = xp + ((long)b * LSEQ + t + 1) * XPD + DTRK + tid;
            unsigned lds = (unsigned)(size_t)(const void*)&sBC[buf ^ 1][tid];
            asm volatile("global_load_async_to_lds_b32 %0, %1, off"
                         :: "v"(lds), "v"(g) : "memory");
            asm volatile("s_wait_asynccnt 0x1" ::: "memory");
        } else {
            asm volatile("s_wait_asynccnt 0x0" ::: "memory");
        }
        __syncthreads();

        long bl = (long)b * LSEQ + t;
        float dv = delta[bl * DINN + d];
        float xv = xbf[bl * DINN + d];
        float acc = 0.f;
#pragma unroll
        for (int n = 0; n < NST; ++n) {
            float dA = __expf(dv * an[n]);
            hst[n] = dA * hst[n] + dv * sBC[buf][n] * xv;
            acc += hst[n] * sBC[buf][NST + n];
        }
        y[bl * DINN + d] = acc + xv * dpar;
        __syncthreads();
    }
}

__global__ void k_gate_bf16(const float* __restrict__ y,
                            const float* __restrict__ xz,
                            __hip_bfloat16* __restrict__ yb) {
    long idx = (long)blockIdx.x * blockDim.x + threadIdx.x;
    if (idx >= (long)BB * LSEQ * DINN) return;
    int d   = (int)(idx % DINN);
    long bl = idx / DINN;
    float z = xz[bl * (2L * DINN) + DINN + d];
    float g = z / (1.f + __expf(-z));
    yb[idx] = __float2bfloat16(y[idx] * g);
}

__global__ void k_add(float* __restrict__ h, const float* __restrict__ p, long n) {
    long i = (long)blockIdx.x * blockDim.x + threadIdx.x;
    if (i < n) h[i] += p[i];
}

__global__ void k_head(const float* __restrict__ hf,
                       const float* __restrict__ hw,
                       float* __restrict__ out) {
    __shared__ float red[256];
    int p = blockIdx.x, b = blockIdx.y, tid = threadIdx.x;
    const float* xr = hf + ((long)b * LSEQ + (LSEQ - 1)) * DMOD;
    const float* wr = hw + (long)p * DMOD;
    float s = 0.f;
    for (int i = tid; i < DMOD; i += blockDim.x) s += xr[i] * wr[i];
    red[tid] = s; __syncthreads();
    for (int o = 128; o; o >>= 1) { if (tid < o) red[tid] += red[tid + o]; __syncthreads(); }
    if (tid == 0) out[(long)b * PHEAD + p] = red[0];
}

// ------------------------------------------------------------------- driver
extern "C" void kernel_launch(void* const* d_in, const int* in_sizes, int n_in,
                              void* d_out, int out_size, void* d_ws, size_t ws_size,
                              hipStream_t stream) {
    const int*   tokens     = (const int*)  d_in[0];
    const float* tok_emb    = (const float*)d_in[1];
    const float* pos_emb    = (const float*)d_in[2];
    const float* ln_w       = (const float*)d_in[3];
    const float* ln_b       = (const float*)d_in[4];
    const float* in_proj_w  = (const float*)d_in[5];
    const float* conv_w     = (const float*)d_in[6];
    const float* conv_b     = (const float*)d_in[7];
    const float* xproj_w    = (const float*)d_in[8];
    const float* dt_w       = (const float*)d_in[9];
    const float* dt_b       = (const float*)d_in[10];
    const float* A_log      = (const float*)d_in[11];
    const float* D_param    = (const float*)d_in[12];
    const float* out_proj_w = (const float*)d_in[13];
    const float* fnorm_w    = (const float*)d_in[14];
    const float* fnorm_b    = (const float*)d_in[15];
    const float* head_w     = (const float*)d_in[16];
    float* out = (float*)d_out;

    const long BL = (long)BB * LSEQ;     // 2048
    char* ws = (char*)d_ws;
    size_t off = 0;
    auto alloc = [&](size_t bytes) -> void* {
        void* p = ws + off;
        off = (off + bytes + 255) & ~(size_t)255;
        return p;
    };

    __hip_bfloat16* wb_ip = (__hip_bfloat16*)alloc((size_t)NLAY * 2 * DINN * DMOD * 2);
    __hip_bfloat16* wb_xp = (__hip_bfloat16*)alloc((size_t)NLAY * XPD * DINN * 2);
    __hip_bfloat16* wb_dt = (__hip_bfloat16*)alloc((size_t)NLAY * DINN * DTRK * 2);
    __hip_bfloat16* wb_op = (__hip_bfloat16*)alloc((size_t)NLAY * DMOD * DINN * 2);
    float*          h     = (float*)         alloc((size_t)BL * DMOD * 4);
    __hip_bfloat16* xln   = (__hip_bfloat16*)alloc((size_t)BL * DMOD * 2);
    float*          xz    = (float*)         alloc((size_t)BL * 2 * DINN * 4);
    float*          xbf   = (float*)         alloc((size_t)BL * DINN * 4);
    __hip_bfloat16* xbb   = (__hip_bfloat16*)alloc((size_t)BL * DINN * 2);
    float*          xp    = (float*)         alloc((size_t)BL * XPD * 4);
    __hip_bfloat16* xpb   = (__hip_bfloat16*)alloc((size_t)BL * XPD * 2);
    float*          delta = (float*)         alloc((size_t)BL * DINN * 4);
    float*          ybuf  = (float*)         alloc((size_t)BL * DINN * 4);
    __hip_bfloat16* ybb   = (__hip_bfloat16*)alloc((size_t)BL * DINN * 2);
    float*          po    = (float*)         alloc((size_t)BL * DMOD * 4);
    float*          hf    = po;              // reuse: po is dead after last residual

    // one-time (per-call) weight conversion to bf16
    k_f32_to_bf16<<<4096, 256, 0, stream>>>(in_proj_w,  wb_ip, (long)NLAY * 2 * DINN * DMOD);
    k_f32_to_bf16<<<4096, 256, 0, stream>>>(xproj_w,    wb_xp, (long)NLAY * XPD * DINN);
    k_f32_to_bf16<<<4096, 256, 0, stream>>>(dt_w,       wb_dt, (long)NLAY * DINN * DTRK);
    k_f32_to_bf16<<<4096, 256, 0, stream>>>(out_proj_w, wb_op, (long)NLAY * DMOD * DINN);

    k_embed<<<(int)BL, 256, 0, stream>>>(tokens, tok_emb, pos_emb, h);

    const dim3 gblk(32, 4);          // 4 wave32 per block; each wave: 16x64 tile
    const int  tilesM = (int)(BL / 16);               // 128
    auto gemmGridX = [](int N) { return ((N >> 4) + 15) / 16; }; // 16 N-tiles/block

    for (int i = 0; i < NLAY; ++i) {
        k_layernorm<<<(int)BL, 256, 0, stream>>>(
            h, ln_w + (size_t)i * DMOD, ln_b + (size_t)i * DMOD,
            (float*)nullptr, xln, DMOD);

        // xz = ln(h) @ in_proj_w^T   (M=2048, N=4096, K=1024)
        k_gemm_bf16<<<dim3(gemmGridX(2 * DINN), tilesM), gblk, 0, stream>>>(
            xln, DMOD, wb_ip + (size_t)i * 2 * DINN * DMOD, DMOD,
            xz, 2 * DINN, (int)BL, 2 * DINN, DMOD);

        k_conv_silu<<<(int)((BL * DINN + 255) / 256), 256, 0, stream>>>(
            xz, conv_w + (size_t)i * DINN * KCV, conv_b + (size_t)i * DINN, xbf, xbb);

        // xp = x @ xproj_w^T   (M=2048, N=160, K=2048)
        k_gemm_bf16<<<dim3(gemmGridX(XPD), tilesM), gblk, 0, stream>>>(
            xbb, DINN, wb_xp + (size_t)i * XPD * DINN, DINN,
            xp, XPD, (int)BL, XPD, DINN);

        k_f32_to_bf16<<<1280, 256, 0, stream>>>(xp, xpb, BL * XPD);

        // delta_pre = dt_r @ dt_w^T  (M=2048, N=2048, K=128; A has lda=160)
        k_gemm_bf16<<<dim3(gemmGridX(DINN), tilesM), gblk, 0, stream>>>(
            xpb, XPD, wb_dt + (size_t)i * DINN * DTRK, DTRK,
            delta, DINN, (int)BL, DINN, DTRK);

        k_softplus_bias<<<(int)((BL * DINN + 255) / 256), 256, 0, stream>>>(
            delta, dt_b + (size_t)i * DINN, BL * DINN, DINN);

        k_ssm_scan<<<dim3(DINN / 256, BB), 256, 0, stream>>>(
            delta, xbf, xp, A_log + (size_t)i * DINN * NST,
            D_param + (size_t)i * DINN, ybuf);

        k_gate_bf16<<<(int)((BL * DINN + 255) / 256), 256, 0, stream>>>(ybuf, xz, ybb);

        // proj = y @ out_proj_w^T  (M=2048, N=1024, K=2048)
        k_gemm_bf16<<<dim3(gemmGridX(DMOD), tilesM), gblk, 0, stream>>>(
            ybb, DINN, wb_op + (size_t)i * DMOD * DINN, DINN,
            po, DMOD, (int)BL, DMOD, DINN);

        k_add<<<(int)((BL * DMOD + 255) / 256), 256, 0, stream>>>(h, po, BL * DMOD);
    }

    k_layernorm<<<(int)BL, 256, 0, stream>>>(
        h, fnorm_w, fnorm_b, hf, (__hip_bfloat16*)nullptr, DMOD);

    k_head<<<dim3(PHEAD, BB), 256, 0, stream>>>(hf, head_w, out);
}
